// SkipGramWordEmbeddings_16071767622088
// MI455X (gfx1250) — compile-verified, matrix operand order checked by reading, not run on hardware
//
#include <hip/hip_runtime.h>
#include <math.h>

// ---------------------------------------------------------------------------
// SkipGram negative-sampling loss, MI455X (gfx1250, wave32).
//
//   loss_b = softplus(-u.pos) + softplus(u . sum_k neg_k)     (log-sigmoid id.)
//   out    = mean_b loss_b
//
// Memory-bound gather kernel (~184 MB gathered reads, ~92 MFLOP).
// - one wave per block; 32 lanes x float4 == one 512B embedding row (coalesced)
// - row indices are wave-uniform -> compiler emits scalar (SMEM) loads
// - dot reductions done with v_wmma_f32_16x16x32_f16 via the ones-matrix
//   trick: A = ones(16x32)  =>  C[m,n] = sum_k B[k,n]; B columns carry the
//   elementwise products of 16 batch rows (staged in padded LDS), 4 K-chunks
//   cover D=128. Products are scaled by 256 before f16 conversion to stay in
//   f16 normal range; the f32 accumulator is unscaled afterwards.
// - deterministic two-stage reduction (block partials in d_ws, tree reduce)
// ---------------------------------------------------------------------------

typedef __attribute__((ext_vector_type(16))) _Float16 v16h;
typedef __attribute__((ext_vector_type(8)))  float    v8f;

#define DIM        128
#define K_NEG      20
#define GROUP      16          // batch rows per block (one WMMA tile width)
#define LDS_STRIDE 132         // 128 + 4 pad: row r starts at bank (4*r)%64 -> conflict-free
#define SCALE      256.0f      // keep elementwise products in f16 normal range
#define INV_SCALE  (1.0f / 256.0f)

__device__ __forceinline__ float softplus_stable(float x) {
    // log(1 + e^x), stable for large |x|
    return fmaxf(x, 0.0f) + log1pf(expf(-fabsf(x)));
}

__global__ __launch_bounds__(32)
void sg_loss_kernel(const float* __restrict__ U_w,
                    const float* __restrict__ V_w,
                    const int*   __restrict__ u_idx,
                    const int*   __restrict__ pos_idx,
                    const int*   __restrict__ neg_idx,
                    float*       __restrict__ partial)
{
    __shared__ __align__(16) float prodp[GROUP * LDS_STRIDE]; // scale * u*pos
    __shared__ __align__(16) float prodn[GROUP * LDS_STRIDE]; // scale * u*sum(neg)

    const int lane = threadIdx.x;        // 0..31 (wave32)
    const int b0   = blockIdx.x * GROUP;

    // ---- gather + stage phase: 22 coalesced float4 row-loads per batch row ----
    for (int i = 0; i < GROUP; ++i) {
        const int b  = b0 + i;
        const int iu = u_idx[b];                         // uniform -> s_load
        const int ip = pos_idx[b];

        const float4 uv = *(const float4*)(U_w + (size_t)iu * DIM + lane * 4);
        const float4 pv = *(const float4*)(V_w + (size_t)ip * DIM + lane * 4);

        float4 ns = make_float4(0.f, 0.f, 0.f, 0.f);
        #pragma unroll
        for (int k = 0; k < K_NEG; ++k) {
            const int in = neg_idx[b * K_NEG + k];       // uniform -> s_load
            const float4 nv = *(const float4*)(V_w + (size_t)in * DIM + lane * 4);
            ns.x += nv.x; ns.y += nv.y; ns.z += nv.z; ns.w += nv.w;
        }

        float4 pp, pn;
        pp.x = uv.x * pv.x * SCALE;  pp.y = uv.y * pv.y * SCALE;
        pp.z = uv.z * pv.z * SCALE;  pp.w = uv.w * pv.w * SCALE;
        pn.x = uv.x * ns.x * SCALE;  pn.y = uv.y * ns.y * SCALE;
        pn.z = uv.z * ns.z * SCALE;  pn.w = uv.w * ns.w * SCALE;

        *(float4*)(&prodp[i * LDS_STRIDE + lane * 4]) = pp;
        *(float4*)(&prodn[i * LDS_STRIDE + lane * 4]) = pn;
    }
    __syncthreads();

    // ---- WMMA reduction phase ----
    // A = ones(16x32) f16: layout-independent (every element identical).
    v16h aones;
    #pragma unroll
    for (int e = 0; e < 16; ++e) aones[e] = (_Float16)1.0f;

    v8f accp = {};   // C for pos dots
    v8f accn = {};   // C for neg dots

    const int n  = lane & 15;          // this lane's B-matrix column (batch row)
    const int hi = (lane >> 4) & 1;    // half-wave selects K subrange (+8)

    #pragma unroll
    for (int c = 0; c < 4; ++c) {      // 4 chunks of K=32 cover D=128
        const int d0 = 32 * c;
        v16h bp, bn;
        #pragma unroll
        for (int e = 0; e < 16; ++e) {
            // 16-bit operand layout (dual of documented A 16x32 layout):
            // VGPR j holds K = 16*(j/4) + 2*(j%4) + {0,1} (+8 for lanes 16..31)
            const int j = e >> 1, h = e & 1;
            const int k = ((j >> 2) << 4) + ((j & 3) << 1) + h + (hi << 3);
            bp[e] = (_Float16)prodp[n * LDS_STRIDE + d0 + k];
            bn[e] = (_Float16)prodn[n * LDS_STRIDE + d0 + k];
        }
        accp = __builtin_amdgcn_wmma_f32_16x16x32_f16(
                   false, aones, false, bp, (short)0, accp, false, false);
        accn = __builtin_amdgcn_wmma_f32_16x16x32_f16(
                   false, aones, false, bn, (short)0, accn, false, false);
    }

    // C VGPR0: lanes 0-15 hold row M=0 (all rows equal under A=ones), N=lane.
    const float s_pos = accp[0] * INV_SCALE;
    const float s_neg = accn[0] * INV_SCALE;

    float loss = softplus_stable(-s_pos) + softplus_stable(s_neg);
    loss = (lane < 16) ? loss : 0.0f;   // lanes 16-31 duplicate (row M=8)

    // wave32 butterfly reduce -> lane 0 holds block sum of 16 losses
    #pragma unroll
    for (int off = 16; off > 0; off >>= 1)
        loss += __shfl_xor(loss, off, 32);

    if (lane == 0) partial[blockIdx.x] = loss;
}

__global__ __launch_bounds__(256)
void sg_reduce_kernel(const float* __restrict__ partial, int n,
                      float* __restrict__ out, float inv_B)
{
    __shared__ float sm[256];
    float s = 0.0f;
    for (int i = threadIdx.x; i < n; i += 256) s += partial[i];
    sm[threadIdx.x] = s;
    __syncthreads();
    #pragma unroll
    for (int stride = 128; stride > 0; stride >>= 1) {
        if ((int)threadIdx.x < stride) sm[threadIdx.x] += sm[threadIdx.x + stride];
        __syncthreads();
    }
    if (threadIdx.x == 0) out[0] = sm[0] * inv_B;
}

extern "C" void kernel_launch(void* const* d_in, const int* in_sizes, int n_in,
                              void* d_out, int out_size, void* d_ws, size_t ws_size,
                              hipStream_t stream)
{
    const float* U_w = (const float*)d_in[0];   // [VOCAB, 128] f32
    const float* V_w = (const float*)d_in[1];   // [VOCAB, 128] f32
    const int*   u   = (const int*)d_in[2];     // [B]
    const int*   pos = (const int*)d_in[3];     // [B]
    const int*   neg = (const int*)d_in[4];     // [B, 20]

    float* out     = (float*)d_out;
    float* partial = (float*)d_ws;              // one f32 per block

    const int B       = in_sizes[2];            // 16384
    const int nblocks = B / GROUP;              // 1024

    sg_loss_kernel<<<nblocks, 32, 0, stream>>>(U_w, V_w, u, pos, neg, partial);
    sg_reduce_kernel<<<1, 256, 0, stream>>>(partial, nblocks, out,
                                            1.0f / (float)B);
}
